// Recommender_36636071035105
// MI455X (gfx1250) — matrix operand order repaired and checked by reading, not compiled, not run on hardware
//
#include <hip/hip_runtime.h>
#include <hip/hip_bf16.h>
#include <math.h>

#define N_USERS 50000
#define N_ITEMS 50000
#define N_NODES 110000
#define EMB 100
#define NF 8
#define META 9
#define N_EDGES 1000000
#define NP3 150000
#define NP5 80000
#define NNZ 1000000
#define BATCH 8192
#define TEMP 0.005f
#define DECAY 1e-5f

#define KPAD 128      // K padded for WMMA (4 x K=32 steps)
#define BSTRIDE 136   // LDS row stride in halves for B tile (bank-conflict pad)

typedef __attribute__((ext_vector_type(16))) _Float16 v16h;
typedef __attribute__((ext_vector_type(8)))  float    v8f;

__device__ __forceinline__ float wave_sum(float v) {
  for (int o = 16; o > 0; o >>= 1) v += __shfl_xor(v, o, 32);
  return v;
}

// ---------------------------------------------------------------------------
// K0: zero-padded f32 -> f16 conversion: [N_USERS][EMB] -> [N_USERS][KPAD]
// ---------------------------------------------------------------------------
__global__ void k_cvt_pad(const float* __restrict__ src,
                          _Float16* __restrict__ dst) {
  long i = (long)blockIdx.x * blockDim.x + threadIdx.x;
  if (i >= (long)N_USERS * KPAD) return;
  int u = (int)(i >> 7);
  int k = (int)(i & (KPAD - 1));
  dst[i] = (_Float16)((k < EMB) ? src[(long)u * EMB + k] : 0.f);
}

// ---------------------------------------------------------------------------
// K1: item_agg scatter: sum all_embed[tails] per head + counts (1 wave/edge)
// ---------------------------------------------------------------------------
__global__ void k_item_scatter(const int* __restrict__ heads,
                               const int* __restrict__ tails,
                               const float* __restrict__ all_embed,
                               float* __restrict__ item_agg,
                               float* __restrict__ cnts) {
  int e = (blockIdx.x * blockDim.x + threadIdx.x) >> 5;
  int lane = threadIdx.x & 31;
  if (e >= N_EDGES) return;
  int h = heads[e];
  long t = tails[e];
  const float* src = all_embed + t * (long)EMB;
  float* dst = item_agg + (long)h * EMB;
  for (int d = lane; d < EMB; d += 32) atomicAdd(&dst[d], src[d]);
  if (lane == 0) atomicAdd(&cnts[h], 1.0f);
}

// ---------------------------------------------------------------------------
// K2: item mean then l2norm in place (1 wave/item)
// ---------------------------------------------------------------------------
__global__ void k_item_finalize(float* __restrict__ item_agg,
                                const float* __restrict__ cnts) {
  int i = (blockIdx.x * blockDim.x + threadIdx.x) >> 5;
  int lane = threadIdx.x & 31;
  if (i >= N_ITEMS) return;
  float c = cnts[i];
  float inv = c > 0.f ? 1.f / fmaxf(c, 1.f) : 0.f;
  float* row = item_agg + (long)i * EMB;
  float v[4]; float ss = 0.f;
  for (int j = 0; j < 4; ++j) {
    int d = lane + 32 * j;
    v[j] = (d < EMB) ? row[d] * inv : 0.f;
    ss += v[j] * v[j];
  }
  ss = wave_sum(ss);
  float s = 1.f / fmaxf(sqrtf(ss), 1e-12f);
  for (int j = 0; j < 4; ++j) {
    int d = lane + 32 * j;
    if (d < EMB) row[d] = v[j] * s;
  }
}

// ---------------------------------------------------------------------------
// K3: fused paths: mean -> score vs latent -> softmax(/TEMP) -> latent_agg
// (1 wave/path, grid-stride; block-local LDS accumulator, then global atomics)
// ---------------------------------------------------------------------------
__global__ void k_paths(const int* __restrict__ p3, const int* __restrict__ p5,
                        const float* __restrict__ all_embed,
                        const float* __restrict__ latent,
                        float* __restrict__ latent_agg_g) {
  __shared__ float sLat[NF * EMB];
  __shared__ float sAcc[NF * EMB];
  for (int i = threadIdx.x; i < NF * EMB; i += blockDim.x) {
    sLat[i] = latent[i];
    sAcc[i] = 0.f;
  }
  __syncthreads();
  int lane = threadIdx.x & 31;
  int gw = (blockIdx.x * blockDim.x + threadIdx.x) >> 5;
  int nWaves = (gridDim.x * blockDim.x) >> 5;
  const int NPATH = NP3 + NP5;
  for (int p = gw; p < NPATH; p += nWaves) {
    float v[4] = {0.f, 0.f, 0.f, 0.f};
    int cnt; const int* nodes;
    if (p < NP3) { cnt = 3; nodes = p3 + (long)p * 3; }
    else         { cnt = 5; nodes = p5 + (long)(p - NP3) * 5; }
    for (int k = 0; k < cnt; ++k) {
      const float* row = all_embed + (long)nodes[k] * EMB;
      for (int j = 0; j < 4; ++j) { int d = lane + 32 * j; if (d < EMB) v[j] += row[d]; }
    }
    float invc = 1.f / (float)cnt;
    for (int j = 0; j < 4; ++j) v[j] *= invc;
    float sc[NF];
    for (int f = 0; f < NF; ++f) {
      float part = 0.f;
      for (int j = 0; j < 4; ++j) { int d = lane + 32 * j; if (d < EMB) part += v[j] * sLat[f * EMB + d]; }
      sc[f] = wave_sum(part);
    }
    float m = sc[0];
    for (int f = 1; f < NF; ++f) m = fmaxf(m, sc[f]);
    float se = 0.f;
    for (int f = 0; f < NF; ++f) { sc[f] = __expf((sc[f] - m) / TEMP); se += sc[f]; }
    float inv = 1.f / se;
    for (int f = 0; f < NF; ++f) {
      float w = sc[f] * inv;
      for (int j = 0; j < 4; ++j) {
        int d = lane + 32 * j;
        if (d < EMB) atomicAdd(&sAcc[f * EMB + d], w * v[j]);
      }
    }
  }
  __syncthreads();
  for (int i = threadIdx.x; i < NF * EMB; i += blockDim.x)
    atomicAdd(&latent_agg_g[i], sAcc[i]);
}

// ---------------------------------------------------------------------------
// K4: disen = l2norm(0.4*softmax(att)@weight + 0.6*latent_agg)  (1 wave/factor)
// ---------------------------------------------------------------------------
__global__ void k_disen(const float* __restrict__ att,
                        const float* __restrict__ weight,
                        const float* __restrict__ latent_agg,
                        float* __restrict__ disen) {
  int f = threadIdx.x >> 5;
  int lane = threadIdx.x & 31;
  if (f >= NF) return;
  float a[META]; float m = -1e30f;
  for (int j = 0; j < META; ++j) { a[j] = att[f * META + j]; m = fmaxf(m, a[j]); }
  float s = 0.f;
  for (int j = 0; j < META; ++j) { a[j] = __expf(a[j] - m); s += a[j]; }
  float inv = 1.f / s;
  float v[4]; float ss = 0.f;
  for (int j = 0; j < 4; ++j) {
    int d = lane + 32 * j; float acc = 0.f;
    if (d < EMB) {
      for (int mm = 0; mm < META; ++mm) acc += a[mm] * inv * weight[mm * EMB + d];
      acc = 0.4f * acc + 0.6f * latent_agg[f * EMB + d];
    }
    v[j] = acc; ss += acc * acc;
  }
  ss = wave_sum(ss);
  float sc = 1.f / fmaxf(sqrtf(ss), 1e-12f);
  for (int j = 0; j < 4; ++j) {
    int d = lane + 32 * j;
    if (d < EMB) disen[f * EMB + d] = v[j] * sc;
  }
}

// ---------------------------------------------------------------------------
// K5: score = softmax(uemb @ latent.T) via v_wmma_f32_16x16x32_f16
// A comes pre-padded f16 [N_USERS][KPAD]: per lane two contiguous 16B chunks
// per K-step (global_load_b128, no branches).
// A layout (16-bit A 16x32, contiguous K): lanes0-15 row=lane, K={0..15};
// lanes16-31 row=lane-16, K={16..31}. B staged in LDS (stride-padded).
// C/D: VGPR r -> rows r (lanes0-15) and r+8 (lanes16-31), N = lane&15.
// ---------------------------------------------------------------------------
__global__ void k_score(const _Float16* __restrict__ Ah,
                        const float* __restrict__ latent,
                        float* __restrict__ score) {
  __shared__ _Float16 sB[16 * BSTRIDE];
  for (int i = threadIdx.x; i < 16 * KPAD; i += blockDim.x) {
    int n = i >> 7, k = i & (KPAD - 1);
    float val = (n < NF && k < EMB) ? latent[n * EMB + k] : 0.f;
    sB[n * BSTRIDE + k] = (_Float16)val;
  }
  __syncthreads();
  int lane = threadIdx.x & 31;
  int tile = (blockIdx.x * blockDim.x + threadIdx.x) >> 5;
  const int nTiles = (N_USERS + 15) / 16;  // 3125, exact
  if (tile >= nTiles) return;
  int row = tile * 16 + (lane & 15);
  const _Float16* A = Ah + (long)row * KPAD;
  int hi = (lane >= 16) ? 16 : 0;
  int n  = lane & 15;
  v8f c = {0.f, 0.f, 0.f, 0.f, 0.f, 0.f, 0.f, 0.f};
#pragma unroll
  for (int ko = 0; ko < KPAD; ko += 32) {
    v16h a, b;
#pragma unroll
    for (int j = 0; j < 8; ++j) {
      a[j]     = A[ko + hi + j];
      a[j + 8] = A[ko + hi + 8 + j];
    }
    int kb = ko + ((lane >= 16) ? 16 : 0);
#pragma unroll
    for (int j = 0; j < 16; ++j) b[j] = sB[n * BSTRIDE + kb + j];
    c = __builtin_amdgcn_wmma_f32_16x16x32_f16(false, a, false, b,
                                               (short)0, c, false, false);
  }
  // softmax over factors N=0..7: xor-shuffle over 8-lane groups
  bool act = (lane & 15) < 8;
#pragma unroll
  for (int r = 0; r < 8; ++r) {
    float v = c[r];
    float mx = v;
    for (int o = 4; o > 0; o >>= 1) mx = fmaxf(mx, __shfl_xor(mx, o, 32));
    float e = __expf(v - mx);
    float s = e;
    for (int o = 4; o > 0; o >>= 1) s += __shfl_xor(s, o, 32);
    float ns = e / s;
    int m = tile * 16 + r + ((lane >= 16) ? 8 : 0);
    if (act) score[(long)m * NF + (lane & 7)] = ns;
  }
}

// ---------------------------------------------------------------------------
// K6: user_agg scatter: vals * item_src[cols] summed per row (1 wave/nnz)
// ---------------------------------------------------------------------------
__global__ void k_user_scatter(const int* __restrict__ rows,
                               const int* __restrict__ cols,
                               const float* __restrict__ vals,
                               const float* __restrict__ item_src,
                               float* __restrict__ user_agg) {
  int e = (blockIdx.x * blockDim.x + threadIdx.x) >> 5;
  int lane = threadIdx.x & 31;
  if (e >= NNZ) return;
  int r = rows[e];
  long ci = cols[e];
  float v = vals[e];
  const float* src = item_src + ci * (long)EMB;
  float* dst = user_agg + (long)r * EMB;
  for (int d = lane; d < EMB; d += 32) atomicAdd(&dst[d], v * src[d]);
}

// ---------------------------------------------------------------------------
// K7: user finalize: user_agg*(1+score@disen), l2norm -> user_emb; user_res acc
// ---------------------------------------------------------------------------
__global__ void k_user_finalize(const float* __restrict__ user_agg,
                                const float* __restrict__ score,
                                const float* __restrict__ disen,
                                const float* __restrict__ user_orig,
                                float* __restrict__ user_emb,
                                float* __restrict__ user_res,
                                int hop) {
  __shared__ float sD[NF * EMB];
  for (int i = threadIdx.x; i < NF * EMB; i += blockDim.x) sD[i] = disen[i];
  __syncthreads();
  int u = (blockIdx.x * blockDim.x + threadIdx.x) >> 5;
  int lane = threadIdx.x & 31;
  if (u >= N_USERS) return;
  float sc[NF];
  for (int f = 0; f < NF; ++f) sc[f] = score[(long)u * NF + f];
  const float* ag = user_agg + (long)u * EMB;
  float v[4]; float ss = 0.f;
  for (int j = 0; j < 4; ++j) {
    int d = lane + 32 * j; float x = 0.f;
    if (d < EMB) {
      float g = 0.f;
      for (int f = 0; f < NF; ++f) g += sc[f] * sD[f * EMB + d];
      x = ag[d] * (1.f + g);
    }
    v[j] = x; ss += x * x;
  }
  ss = wave_sum(ss);
  float s = 1.f / fmaxf(sqrtf(ss), 1e-12f);
  for (int j = 0; j < 4; ++j) {
    int d = lane + 32 * j;
    if (d < EMB) {
      float ue = v[j] * s;
      user_emb[(long)u * EMB + d] = ue;
      if (hop == 0) user_res[(long)u * EMB + d] = user_orig[(long)u * EMB + d] + ue;
      else          user_res[(long)u * EMB + d] += ue;
    }
  }
}

// ---------------------------------------------------------------------------
// K8: batch loss reduction (1 wave/sample); item_res = item_orig + 2*item_norm
// ---------------------------------------------------------------------------
__global__ void k_loss(const int* __restrict__ bu, const int* __restrict__ bi,
                       const int* __restrict__ bn,
                       const float* __restrict__ user_res,
                       const float* __restrict__ all_embed,
                       const float* __restrict__ item_norm,
                       float* __restrict__ acc) {
  int b = (blockIdx.x * blockDim.x + threadIdx.x) >> 5;
  int lane = threadIdx.x & 31;
  if (b >= BATCH) return;
  int u = bu[b];
  int ip = bi[b] - N_USERS;
  int in = bn[b];
  const float* ue = user_res + (long)u * EMB;
  const float* po = all_embed + (long)(N_USERS + ip) * EMB;
  const float* pn = item_norm + (long)ip * EMB;
  const float* no = all_embed + (long)(N_USERS + in) * EMB;
  const float* nn = item_norm + (long)in * EMB;
  float ps = 0.f, ns = 0.f, reg = 0.f;
  for (int d = lane; d < EMB; d += 32) {
    float uu = ue[d];
    float pe = po[d] + 2.f * pn[d];
    float ne = no[d] + 2.f * nn[d];
    ps += uu * pe; ns += uu * ne;
    reg += uu * uu + pe * pe + ne * ne;
  }
  ps = wave_sum(ps); ns = wave_sum(ns); reg = wave_sum(reg);
  if (lane == 0) {
    float x = ps - ns;
    float lsig = fminf(x, 0.f) - log1pf(__expf(-fabsf(x)));
    atomicAdd(&acc[0], lsig);
    atomicAdd(&acc[1], reg);
  }
}

__global__ void k_final(const float* __restrict__ acc, float* __restrict__ out) {
  if (threadIdx.x == 0 && blockIdx.x == 0) {
    float mf  = -acc[0] / (float)BATCH;
    float emb = DECAY * (acc[1] * 0.5f) / (float)BATCH;
    out[0] = mf + emb;
    out[1] = mf;
    out[2] = emb;
  }
}

// ---------------------------------------------------------------------------
extern "C" void kernel_launch(void* const* d_in, const int* in_sizes, int n_in,
                              void* d_out, int out_size, void* d_ws, size_t ws_size,
                              hipStream_t stream) {
  (void)in_sizes; (void)n_in; (void)out_size; (void)ws_size;
  const int* batch_users = (const int*)d_in[0];
  const int* batch_item  = (const int*)d_in[1];
  const int* neg         = (const int*)d_in[2];
  const int* heads_idx   = (const int*)d_in[3];
  const int* tails_idx   = (const int*)d_in[4];
  const int* path3       = (const int*)d_in[5];
  const int* path5       = (const int*)d_in[6];
  const int* mean_rows   = (const int*)d_in[7];
  const int* mean_cols   = (const int*)d_in[8];
  const float* mean_vals = (const float*)d_in[9];
  const float* all_embed = (const float*)d_in[10];
  const float* latent    = (const float*)d_in[11];
  const float* weight    = (const float*)d_in[12];
  const float* att       = (const float*)d_in[13];
  float* out = (float*)d_out;

  float* w = (float*)d_ws;
  size_t o = 0;
  float* item_agg   = w + o; o += (size_t)N_ITEMS * EMB;   // -> item_norm in place
  float* cnts       = w + o; o += N_ITEMS;
  float* user_emb   = w + o; o += (size_t)N_USERS * EMB;
  float* user_agg   = w + o; o += (size_t)N_USERS * EMB;
  float* user_res   = w + o; o += (size_t)N_USERS * EMB;
  float* score      = w + o; o += (size_t)N_USERS * NF;
  float* latent_agg = w + o; o += NF * EMB;
  float* disen      = w + o; o += NF * EMB;
  float* acc        = w + o; o += 2;
  _Float16* uemb_h  = (_Float16*)(w + o); o += ((size_t)N_USERS * KPAD) / 2;

  // zero init (item_agg+cnts contiguous)
  hipMemsetAsync(item_agg, 0, ((size_t)N_ITEMS * EMB + N_ITEMS) * sizeof(float), stream);
  hipMemsetAsync(latent_agg, 0, NF * EMB * sizeof(float), stream);
  hipMemsetAsync(acc, 0, 2 * sizeof(float), stream);

  // item aggregation
  k_item_scatter<<<N_EDGES / 8, 256, 0, stream>>>(heads_idx, tails_idx, all_embed, item_agg, cnts);
  k_item_finalize<<<N_ITEMS / 8, 256, 0, stream>>>(item_agg, cnts);

  // paths -> latent_agg -> disen
  k_paths<<<512, 256, 0, stream>>>(path3, path5, all_embed, latent, latent_agg);
  k_disen<<<1, 256, 0, stream>>>(att, weight, latent_agg, disen);

  const float* item_orig = all_embed + (size_t)N_USERS * EMB;
  const int cvtBlocks = (int)(((size_t)N_USERS * KPAD + 255) / 256);
  const int scoreBlocks = (((N_USERS + 15) / 16) * 32 + 255) / 256;

  // hop 1 (user src = original users, item src = original items)
  hipMemsetAsync(user_agg, 0, (size_t)N_USERS * EMB * sizeof(float), stream);
  k_cvt_pad<<<cvtBlocks, 256, 0, stream>>>(all_embed, uemb_h);
  k_score<<<scoreBlocks, 256, 0, stream>>>(uemb_h, latent, score);
  k_user_scatter<<<NNZ / 8, 256, 0, stream>>>(mean_rows, mean_cols, mean_vals, item_orig, user_agg);
  k_user_finalize<<<N_USERS / 8, 256, 0, stream>>>(user_agg, score, disen, all_embed, user_emb, user_res, 0);

  // hop 2 (user src = user_emb from hop1, item src = l2norm(item_agg))
  hipMemsetAsync(user_agg, 0, (size_t)N_USERS * EMB * sizeof(float), stream);
  k_cvt_pad<<<cvtBlocks, 256, 0, stream>>>(user_emb, uemb_h);
  k_score<<<scoreBlocks, 256, 0, stream>>>(uemb_h, latent, score);
  k_user_scatter<<<NNZ / 8, 256, 0, stream>>>(mean_rows, mean_cols, mean_vals, item_agg, user_agg);
  k_user_finalize<<<N_USERS / 8, 256, 0, stream>>>(user_agg, score, disen, all_embed, user_emb, user_res, 1);

  // loss
  k_loss<<<BATCH / 8, 256, 0, stream>>>(batch_users, batch_item, neg, user_res, all_embed, item_agg, acc);
  k_final<<<1, 1, 0, stream>>>(acc, out);
}
